// Siren_74689481278010
// MI455X (gfx1250) — matrix operand order repaired and checked by reading, not compiled, not run on hardware
//
#include <hip/hip_runtime.h>

typedef __attribute__((ext_vector_type(16))) _Float16 v16h;
typedef __attribute__((ext_vector_type(8)))  _Float16 v8h;
typedef __attribute__((ext_vector_type(4)))  _Float16 v4h;
typedef __attribute__((ext_vector_type(8)))  float    v8f;

#define HID    128
#define NHID   6
#define OMEGA  50.0f
#define ROWS_PER_BLOCK 128   // 8 waves * 16 rows

__global__ __launch_bounds__(256) void siren_wmma_kernel(
    const float* __restrict__ grid,
    const float* __restrict__ W1, const float* __restrict__ b1,
    const float* __restrict__ Wh, const float* __restrict__ bh,
    const float* __restrict__ Wo, const float* __restrict__ bo,
    float* __restrict__ out)
{
    // 32KB weight buffer (f16 [128][128]) + 32KB activation buffer (f16 [8 waves][16][128])
    __shared__ __align__(16) unsigned char smem[65536];
    _Float16* Wlds = (_Float16*)smem;
    _Float16* Xlds = (_Float16*)(smem + 32768);

    const int tid  = threadIdx.x;
    const int wave = tid >> 5;
    const int lane = tid & 31;
    const int l15  = lane & 15;
    const int hi   = lane >> 4;                 // 0 or 1 (half-wave)
    _Float16* Xw = Xlds + wave * (16 * HID);    // this wave's 16-row activation slab
    const int rowbase = blockIdx.x * ROWS_PER_BLOCK + wave * 16;

    // ---- stage W1 (128x2) and b1 into the (currently unused) weight buffer, as f32 ----
    float* W1s = (float*)smem;        // 256 floats
    float* b1s = (float*)smem + 256;  // 128 floats
    W1s[tid] = W1[tid];               // tid in [0,256)
    if (tid < HID) b1s[tid] = b1[tid];
    __syncthreads();

    // ---- first SineLayer: 2 -> 128, pure VALU (K=2 is too small for WMMA) ----
    {
        const int row   = lane >> 1;          // 0..15 within wave's slab
        const int obase = (lane & 1) * 64;    // each lane covers 64 output features
        const int rg    = rowbase + row;
        const float x0  = grid[rg * 2 + 0] * 2.0f - 1.0f;
        const float x1  = grid[rg * 2 + 1] * 2.0f - 1.0f;
        #pragma unroll 8
        for (int o = 0; o < 64; o += 2) {
            const int oo = obase + o;
            float z0 = __sinf(OMEGA * (x0 * W1s[oo * 2 + 0] + x1 * W1s[oo * 2 + 1] + b1s[oo]));
            float z1 = __sinf(OMEGA * (x0 * W1s[(oo + 1) * 2 + 0] + x1 * W1s[(oo + 1) * 2 + 1] + b1s[oo + 1]));
            union { _Float16 h[2]; unsigned u; } p;
            p.h[0] = (_Float16)z0; p.h[1] = (_Float16)z1;
            *(unsigned*)(Xw + row * HID + oo) = p.u;   // packed b32 store to LDS
        }
    }
    __syncthreads();  // W1s reads done; weight buffer may be overwritten

    // ---- six hidden SineLayers: X[16,128] = sin(OMEGA*(X @ W^T + b)) via WMMA ----
    for (int l = 0; l < NHID; ++l) {
        // cooperatively stage W_l (f32 -> f16) into LDS: 16384 elems / 256 threads
        const float* Wl = Wh + l * (HID * HID);
        #pragma unroll
        for (int it = 0; it < 16; ++it) {
            const int idx = tid * 4 + it * 1024;
            float4 w = *(const float4*)(Wl + idx);
            v4h h; h[0] = (_Float16)w.x; h[1] = (_Float16)w.y;
                   h[2] = (_Float16)w.z; h[3] = (_Float16)w.w;
            *(v4h*)(Wlds + idx) = h;
        }
        // pull next layer's weights toward L2 while we compute
        if (l + 1 < NHID)
            __builtin_prefetch(Wh + (l + 1) * (HID * HID) + tid * 64, 0, 1);
        __syncthreads();

        v8f acc[8];
        #pragma unroll
        for (int nb = 0; nb < 8; ++nb)
            #pragma unroll
            for (int j = 0; j < 8; ++j) acc[nb][j] = 0.0f;

        #pragma unroll
        for (int kc = 0; kc < 4; ++kc) {
            // A fragment (16x32 f16): lanes 0-15 -> K kc*32+0..15; lanes 16-31 -> kc*32+16..31
            const _Float16* ap = Xw + l15 * HID + kc * 32 + hi * 16;
            v8h a0 = *(const v8h*)(ap);
            v8h a1 = *(const v8h*)(ap + 8);
            v16h A;
            #pragma unroll
            for (int i = 0; i < 8; ++i) { A[i] = a0[i]; A[i + 8] = a1[i]; }

            #pragma unroll
            for (int nb = 0; nb < 8; ++nb) {
                // B fragment (32x16 f16): lane holds column n = nb*16+l15, 16 contiguous K
                const _Float16* bp = Wlds + (nb * 16 + l15) * HID + kc * 32 + hi * 16;
                v16h B = *(const v16h*)(bp);
                acc[nb] = __builtin_amdgcn_wmma_f32_16x16x32_f16(
                    false, A, false, B, (short)0, acc[nb], false, false);
            }
        }

        // bias + sin + f16 convert, scatter back to wave-private slab (C layout)
        #pragma unroll
        for (int nb = 0; nb < 8; ++nb) {
            const float bv = bh[l * HID + nb * 16 + l15];
            #pragma unroll
            for (int r = 0; r < 8; ++r) {
                float z = __sinf(OMEGA * (acc[nb][r] + bv));
                Xw[(r + hi * 8) * HID + nb * 16 + l15] = (_Float16)z;
            }
        }
        __syncthreads();  // all waves done with Wlds before next stage overwrites it
    }

    // ---- output layer: 128 -> 3, as one WMMA tile with Wo zero-padded to 16 rows ----
    {
        const int row = tid >> 4;          // 0..15
        const int c8  = (tid & 15) * 8;    // 8-float chunk
        v8h h;
        if (row < 3) {
            const float* p = Wo + row * HID + c8;
            #pragma unroll
            for (int i = 0; i < 8; ++i) h[i] = (_Float16)p[i];
        } else {
            #pragma unroll
            for (int i = 0; i < 8; ++i) h[i] = (_Float16)0.0f;
        }
        *(v8h*)(Wlds + row * HID + c8) = h;
    }
    __syncthreads();

    {
        v8f acc;
        #pragma unroll
        for (int j = 0; j < 8; ++j) acc[j] = 0.0f;

        #pragma unroll
        for (int kc = 0; kc < 4; ++kc) {
            const _Float16* ap = Xw + l15 * HID + kc * 32 + hi * 16;
            v8h a0 = *(const v8h*)(ap);
            v8h a1 = *(const v8h*)(ap + 8);
            v16h A;
            #pragma unroll
            for (int i = 0; i < 8; ++i) { A[i] = a0[i]; A[i + 8] = a1[i]; }
            const _Float16* bp = Wlds + l15 * HID + kc * 32 + hi * 16;
            v16h B = *(const v16h*)(bp);
            acc = __builtin_amdgcn_wmma_f32_16x16x32_f16(
                false, A, false, B, (short)0, acc, false, false);
        }

        if (l15 < 3) {                      // only 3 valid output columns
            const float bv = bo[l15];
            #pragma unroll
            for (int r = 0; r < 8; ++r) {
                const int rg = rowbase + r + hi * 8;
                out[rg * 3 + l15] = (acc[r] + bv) * 0.5f + 0.5f;
            }
        }
    }
}

extern "C" void kernel_launch(void* const* d_in, const int* in_sizes, int n_in,
                              void* d_out, int out_size, void* d_ws, size_t ws_size,
                              hipStream_t stream) {
    const float* grid = (const float*)d_in[0];
    const float* W1   = (const float*)d_in[1];
    const float* b1   = (const float*)d_in[2];
    const float* Wh   = (const float*)d_in[3];
    const float* bh   = (const float*)d_in[4];
    const float* Wo   = (const float*)d_in[5];
    const float* bo   = (const float*)d_in[6];
    float* out = (float*)d_out;

    const int n_rows = 1024 * 1024;
    dim3 gridDim(n_rows / ROWS_PER_BLOCK);   // 8192 blocks
    dim3 blockDim(256);                      // 8 wave32s
    siren_wmma_kernel<<<gridDim, blockDim, 0, stream>>>(grid, W1, b1, Wh, bh, Wo, bo, out);
}
